// GroveMoE_38878043964066
// MI455X (gfx1250) — compile-verified, matrix operand order checked by reading, not run on hardware
//
#include <hip/hip_runtime.h>
#include <hip/hip_fp16.h>
#include <math.h>

#define E_      32
#define G_      16
#define HID_    1024
#define INTER_  2048
#define AINTER_ 128
#define T_      4096
#define CAP_    160
#define SCALE_  0.05f

typedef __attribute__((ext_vector_type(16))) _Float16 v16h;
typedef __attribute__((ext_vector_type(8)))  float    v8f;
typedef unsigned int u32;
typedef __attribute__((ext_vector_type(4))) u32 u32x4;
typedef __attribute__((ext_vector_type(4))) int i32x4;
typedef __attribute__((ext_vector_type(8))) int i32x8;

// ---- WMMA fragment helpers (ISA 7.12.2 16-bit operand layout) ----
__device__ __forceinline__ int frag_k(int k0, int lane, int i) {
    int kb = k0 + ((lane & 16) ? 8 : 0);
    return (i < 8) ? (kb + i) : (kb + 8 + i);   // halves 0..7 -> kb..kb+7 ; 8..15 -> kb+16..kb+23
}

__device__ __forceinline__ v16h load_a_f16(const _Float16* A, int lda,
                                           int m0, int k0, int lane) {
    int m = m0 + (lane & 15);
    const _Float16* p = A + (size_t)m * lda;
    v16h a;
#pragma unroll
    for (int i = 0; i < 16; ++i) a[i] = p[frag_k(k0, lane, i)];
    return a;
}

// B[k][n] = W[n][k], W row-major N x K in fp32, convert to f16 inline
__device__ __forceinline__ v16h load_b_f32T(const float* __restrict__ W, int ldw,
                                            int n0, int k0, int lane) {
    int n = n0 + (lane & 15);
    const float* p = W + (size_t)n * ldw;
    v16h b;
#pragma unroll
    for (int i = 0; i < 16; ++i) b[i] = (_Float16)p[frag_k(k0, lane, i)];
    return b;
}

__device__ __forceinline__ v8f wmma_f16(v16h a, v16h b, v8f c) {
    return __builtin_amdgcn_wmma_f32_16x16x32_f16(false, a, false, b, (short)0, c, false, false);
}

// ---- TDM: stage a 2D f16 tile (rows x cols, row stride in elements) into LDS ----
// D# bitfields per ISA 8.3 (group0) / 8.4 (group1).
__device__ __forceinline__ void tdm_load_2d_f16(const _Float16* gsrc, void* lds_dst,
                                                int rows, int cols, int row_stride) {
    u32 lds_off = (u32)(uintptr_t)lds_dst;                  // LDS byte offset (addr[31:0])
    unsigned long long ga = (unsigned long long)(uintptr_t)gsrc;
    u32x4 g0;
    g0[0] = 1u;                                             // count=1, user descriptor
    g0[1] = lds_off;                                        // lds_addr          [63:32]
    g0[2] = (u32)(ga & 0xFFFFFFFFu);                        // global_addr lo    [95:64]
    g0[3] = (u32)((ga >> 32) & 0x01FFFFFFu) | (2u << 30);   // addr hi + type=2  [127:96]
    i32x8 g1;
    g1[0] = (1 << 16);                                      // data_size=1 (2B), mask=0
    g1[1] = (cols & 0xFFFF) << 16;                          // tensor_dim0[15:0]
    g1[2] = ((cols >> 16) & 0xFFFF) | ((rows & 0xFFFF) << 16); // dim0 hi | tensor_dim1 lo
    g1[3] = ((rows >> 16) & 0xFFFF) | ((cols & 0xFFFF) << 16); // dim1 hi | tile_dim0
    g1[4] = rows & 0xFFFF;                                  // tile_dim1, tile_dim2=0
    g1[5] = row_stride;                                     // tensor_dim0_stride lo
    g1[6] = 0;                                              // stride hi | dim1_stride lo
    g1[7] = 0;
    i32x4 z4 = {};
#if __clang_major__ >= 23
    i32x8 z8 = {};
    __builtin_amdgcn_tensor_load_to_lds(g0, g1, z4, z4, z8, 0);
#else
    __builtin_amdgcn_tensor_load_to_lds(g0, g1, z4, z4, 0);
#endif
}

// ---------------- Router: stats -> MLP -> softmax -> top2 ----------------
__global__ void __launch_bounds__(256) router_kernel(
    const float* __restrict__ x, const float* __restrict__ r1_w,
    const float* __restrict__ r1_b, const float* __restrict__ r2_w,
    float* __restrict__ probs, int* __restrict__ i2, float* __restrict__ w2) {
    __shared__ float ri[HID_ + 6];
    __shared__ float hs[512];
    __shared__ float red[256];
    __shared__ float lg[E_];
    int t = blockIdx.x, tid = threadIdx.x;
    const float* xr = x + (size_t)t * HID_;
    for (int i = tid; i < HID_; i += 256) ri[i] = xr[i];
    __syncthreads();

    float s = 0.f, ss = 0.f, mn = 1e30f, mx = -1e30f, zc = 0.f;
    for (int i = tid; i < HID_; i += 256) {
        float v = ri[i];
        s += v; ss += v * v;
        mn = fminf(mn, v); mx = fmaxf(mx, v);
        zc += (fabsf(v) < 1e-6f) ? 1.f : 0.f;
    }
    float sum, sumsq, minv, maxv, zcnt;
    red[tid] = s;  __syncthreads();
    for (int o = 128; o > 0; o >>= 1) { if (tid < o) red[tid] += red[tid + o]; __syncthreads(); }
    sum = red[0]; __syncthreads();
    red[tid] = ss; __syncthreads();
    for (int o = 128; o > 0; o >>= 1) { if (tid < o) red[tid] += red[tid + o]; __syncthreads(); }
    sumsq = red[0]; __syncthreads();
    red[tid] = mn; __syncthreads();
    for (int o = 128; o > 0; o >>= 1) { if (tid < o) red[tid] = fminf(red[tid], red[tid + o]); __syncthreads(); }
    minv = red[0]; __syncthreads();
    red[tid] = mx; __syncthreads();
    for (int o = 128; o > 0; o >>= 1) { if (tid < o) red[tid] = fmaxf(red[tid], red[tid + o]); __syncthreads(); }
    maxv = red[0]; __syncthreads();
    red[tid] = zc; __syncthreads();
    for (int o = 128; o > 0; o >>= 1) { if (tid < o) red[tid] += red[tid + o]; __syncthreads(); }
    zcnt = red[0]; __syncthreads();

    if (tid == 0) {
        float mean = sum / (float)HID_;
        float var  = (sumsq - (float)HID_ * mean * mean) / (float)(HID_ - 1);
        ri[HID_ + 0] = mean;
        ri[HID_ + 1] = sqrtf(fmaxf(var, 0.f));
        ri[HID_ + 2] = minv;
        ri[HID_ + 3] = maxv;
        ri[HID_ + 4] = sqrtf(sumsq);
        ri[HID_ + 5] = zcnt / (float)HID_;
    }
    __syncthreads();

    for (int j = tid; j < 512; j += 256) {
        const float* wr = r1_w + (size_t)j * (HID_ + 6);
        float d = r1_b[j];
        for (int i = 0; i < HID_ + 6; ++i) d += ri[i] * wr[i];
        hs[j] = 0.5f * d * (1.f + erff(d * 0.70710678118654752f));   // exact GELU
    }
    __syncthreads();

    {   // logits: 8 partial lanes per expert
        int e = tid >> 3, p = tid & 7;
        const float* wr = r2_w + (size_t)e * 512 + p * 64;
        const float* hh = hs + p * 64;
        float d = 0.f;
        for (int i = 0; i < 64; ++i) d += hh[i] * wr[i];
        red[tid] = d;
    }
    __syncthreads();
    if (tid < E_) {
        float d = 0.f;
        for (int p = 0; p < 8; ++p) d += red[tid * 8 + p];
        lg[tid] = d;
    }
    __syncthreads();
    if (tid == 0) {
        float m = lg[0];
        for (int e = 1; e < E_; ++e) m = fmaxf(m, lg[e]);
        float den = 0.f;
        for (int e = 0; e < E_; ++e) { lg[e] = expf(lg[e] - m); den += lg[e]; }
        int b0 = 0, b1 = -1; float p0 = -1.f, p1 = -1.f;
        for (int e = 0; e < E_; ++e) {
            float pv = lg[e] / den;
            probs[(size_t)t * E_ + e] = pv;
            if (pv > p0)      { p1 = p0; b1 = b0; p0 = pv; b0 = e; }
            else if (pv > p1) { p1 = pv; b1 = e; }
        }
        float w0 = p0 / (p0 + p1);
        i2[t * 2] = b0;  i2[t * 2 + 1] = b1;
        w2[t * 2] = w0;  w2[t * 2 + 1] = 1.f - w0;
    }
}

// ---------------- Capacity: per-expert stable top-cap selection ----------------
__global__ void __launch_bounds__(256) capacity_kernel(
    const float* __restrict__ probs, const int* __restrict__ i2, const float* __restrict__ w2,
    int* __restrict__ idxcap, float* __restrict__ wcap, float* __restrict__ gw) {
    __shared__ float sc[T_];
    __shared__ float rv[256];
    __shared__ int   rix[256];
    int e = blockIdx.x, tid = threadIdx.x;
    for (int t = tid; t < T_; t += 256) {
        bool routed = (i2[t * 2] == e) || (i2[t * 2 + 1] == e);
        sc[t] = routed ? probs[(size_t)t * E_ + e] : -1e30f;
    }
    __syncthreads();
    for (int s = 0; s < CAP_; ++s) {
        float bv = -2e30f; int bi = T_;
        for (int t = tid; t < T_; t += 256) {
            float v = sc[t];
            if (v > bv || (v == bv && t < bi)) { bv = v; bi = t; }
        }
        rv[tid] = bv; rix[tid] = bi;
        __syncthreads();
        for (int o = 128; o > 0; o >>= 1) {
            if (tid < o) {
                float v2 = rv[tid + o]; int j = rix[tid + o];
                if (v2 > rv[tid] || (v2 == rv[tid] && j < rix[tid])) { rv[tid] = v2; rix[tid] = j; }
            }
            __syncthreads();
        }
        if (tid == 0) {
            int bt = rix[0];
            float wv = 0.f;
            if (i2[bt * 2] == e)          wv = w2[bt * 2];
            else if (i2[bt * 2 + 1] == e) wv = w2[bt * 2 + 1];
            idxcap[e * CAP_ + s] = bt;
            wcap[e * CAP_ + s]   = wv;
            if (wv != 0.f) atomicAdd(&gw[(size_t)bt * G_ + (e >> 1)], SCALE_ * wv);
            sc[bt] = -3e30f;
        }
        __syncthreads();
    }
}

// ---------------- fp32 -> f16 convert / gather ----------------
__global__ void __launch_bounds__(256) cvt_half_kernel(const float* __restrict__ src,
                                                       _Float16* __restrict__ dst, int n) {
    int i = blockIdx.x * 256 + threadIdx.x;
    if (i < n) dst[i] = (_Float16)src[i];
}

__global__ void __launch_bounds__(256) gather_kernel(const float* __restrict__ xf,
                                                     const int* __restrict__ idxcap,
                                                     _Float16* __restrict__ xe) {
    int i = blockIdx.x * 256 + threadIdx.x;       // over E_*CAP_*HID_
    int c = i & (HID_ - 1);
    int row = i >> 10;                            // e*CAP_ + slot
    int tkn = idxcap[row];
    xe[i] = (_Float16)xf[(size_t)tkn * HID_ + c];
}

// ---------------- Expert up-proj + SwiGLU (TDM-staged A, WMMA) ----------------
// block = (expert, m-tile); A panel 16x1024 f16 staged to LDS by TDM; 8 waves sweep 128 j-tiles
__global__ void __launch_bounds__(256) expert_up_kernel(
    const _Float16* __restrict__ xe, const float* __restrict__ w_up,
    _Float16* __restrict__ hact) {
    __shared__ _Float16 atile[16 * HID_];                 // 32 KB
    const int TM = CAP_ / 16;
    int e  = blockIdx.x / TM;
    int m0 = (blockIdx.x % TM) * 16;
    if (threadIdx.x < 32) {
        tdm_load_2d_f16(xe + ((size_t)e * CAP_ + m0) * HID_, atile, 16, HID_, HID_);
        __builtin_amdgcn_s_wait_tensorcnt(0);
    }
    __syncthreads();

    int wid = threadIdx.x >> 5, lane = threadIdx.x & 31;
    const float* Wg = w_up + (size_t)e * 2 * INTER_ * HID_;
    _Float16* H = hact + (size_t)e * CAP_ * INTER_;
    int mrow = m0 + ((lane & 16) ? 8 : 0);
    int nlo  = lane & 15;
    for (int jt = wid; jt < INTER_ / 16; jt += 8) {
        int j0 = jt * 16;
        v8f cg = {}, cu = {};
        for (int k0 = 0; k0 < HID_; k0 += 32) {
            v16h a  = load_a_f16(atile, HID_, 0, k0, lane);       // ds_load from LDS
            v16h bg = load_b_f32T(Wg, HID_, j0, k0, lane);
            v16h bu = load_b_f32T(Wg, HID_, INTER_ + j0, k0, lane);
            cg = wmma_f16(a, bg, cg);
            cu = wmma_f16(a, bu, cu);
        }
#pragma unroll
        for (int rr = 0; rr < 8; ++rr) {
            float g = cg[rr], u = cu[rr];
            float h = (g / (1.f + expf(-g))) * u;                 // silu(gate)*up
            H[(size_t)(mrow + rr) * INTER_ + j0 + nlo] = (_Float16)h;
        }
    }
}

// ---------------- Expert down-proj + weighted scatter (TDM-staged A, WMMA + atomics) ----------------
__global__ void __launch_bounds__(256) expert_down_kernel(
    const _Float16* __restrict__ hact, const float* __restrict__ w_down,
    const int* __restrict__ idxcap, const float* __restrict__ wcap,
    float* __restrict__ out) {
    __shared__ _Float16 atile[16 * INTER_];               // 64 KB
    const int TM = CAP_ / 16;
    int e  = blockIdx.x / TM;
    int m0 = (blockIdx.x % TM) * 16;
    if (threadIdx.x < 32) {
        tdm_load_2d_f16(hact + ((size_t)e * CAP_ + m0) * INTER_, atile, 16, INTER_, INTER_);
        __builtin_amdgcn_s_wait_tensorcnt(0);
    }
    __syncthreads();

    int wid = threadIdx.x >> 5, lane = threadIdx.x & 31;
    const float* W = w_down + (size_t)e * HID_ * INTER_;
    int mrow = m0 + ((lane & 16) ? 8 : 0);
    int nlo  = lane & 15;
    for (int ot = wid; ot < HID_ / 16; ot += 8) {
        int o0 = ot * 16;
        v8f c = {};
        for (int k0 = 0; k0 < INTER_; k0 += 32) {
            v16h a = load_a_f16(atile, INTER_, 0, k0, lane);
            v16h b = load_b_f32T(W, INTER_, o0, k0, lane);
            c = wmma_f16(a, b, c);
        }
#pragma unroll
        for (int rr = 0; rr < 8; ++rr) {
            int slot = mrow + rr;
            float wv = wcap[e * CAP_ + slot];
            if (wv != 0.f) {
                int tkn = idxcap[e * CAP_ + slot];
                atomicAdd(&out[(size_t)tkn * HID_ + o0 + nlo], wv * c[rr]);
            }
        }
    }
}

// ---------------- Adjugate up-proj + SwiGLU, gw folded (TDM-staged A, WMMA) ----------------
// block = (group, token-tile); 8 waves cover the 8 j-tiles of one group exactly
__global__ void __launch_bounds__(256) adj_up_kernel(
    const _Float16* __restrict__ xh, const float* __restrict__ a_up,
    const float* __restrict__ gw, _Float16* __restrict__ ahw) {
    __shared__ _Float16 atile[16 * HID_];                 // 32 KB
    const int TM = T_ / 16;
    int g  = blockIdx.x / TM;
    int m0 = (blockIdx.x % TM) * 16;
    if (threadIdx.x < 32) {
        tdm_load_2d_f16(xh + (size_t)m0 * HID_, atile, 16, HID_, HID_);
        __builtin_amdgcn_s_wait_tensorcnt(0);
    }
    __syncthreads();

    int wid = threadIdx.x >> 5, lane = threadIdx.x & 31;
    const float* Wg = a_up + (size_t)g * 2 * AINTER_ * HID_;
    int j0 = wid * 16;                                    // AINTER_/16 == 8 == waves/block
    v8f cg = {}, cu = {};
    for (int k0 = 0; k0 < HID_; k0 += 32) {
        v16h a  = load_a_f16(atile, HID_, 0, k0, lane);
        v16h bg = load_b_f32T(Wg, HID_, j0, k0, lane);
        v16h bu = load_b_f32T(Wg, HID_, AINTER_ + j0, k0, lane);
        cg = wmma_f16(a, bg, cg);
        cu = wmma_f16(a, bu, cu);
    }
    int mrow = m0 + ((lane & 16) ? 8 : 0);
    int col  = j0 + (lane & 15);
#pragma unroll
    for (int rr = 0; rr < 8; ++rr) {
        int t = mrow + rr;
        float gwv = gw[(size_t)t * G_ + g];
        float gg = cg[rr], u = cu[rr];
        float h = (gg / (1.f + expf(-gg))) * u * gwv;
        ahw[(size_t)t * (G_ * AINTER_) + g * AINTER_ + col] = (_Float16)h;
    }
}

// ---------------- Adjugate down-proj: dense [T,2048]x[2048,HID] GEMM (TDM-staged A) ----------------
__global__ void __launch_bounds__(256) adj_down_kernel(
    const _Float16* __restrict__ ahw, const float* __restrict__ a_down,
    float* __restrict__ out) {
    const int K = G_ * AINTER_;                           // 2048
    __shared__ _Float16 atile[16 * K];                    // 64 KB
    int m0 = blockIdx.x * 16;
    if (threadIdx.x < 32) {
        tdm_load_2d_f16(ahw + (size_t)m0 * K, atile, 16, K, K);
        __builtin_amdgcn_s_wait_tensorcnt(0);
    }
    __syncthreads();

    int wid = threadIdx.x >> 5, lane = threadIdx.x & 31;
    int mrow = m0 + ((lane & 16) ? 8 : 0);
    int nlo  = lane & 15;
    for (int ot = wid; ot < HID_ / 16; ot += 8) {
        int o0 = ot * 16;
        int n  = o0 + nlo;
        v8f c = {};
        for (int k0 = 0; k0 < K; k0 += 32) {
            v16h a = load_a_f16(atile, K, 0, k0, lane);
            v16h b;
#pragma unroll
            for (int i = 0; i < 16; ++i) {
                int k = frag_k(k0, lane, i);
                int g = k >> 7, j = k & (AINTER_ - 1);
                b[i] = (_Float16)a_down[((size_t)g * HID_ + n) * AINTER_ + j];
            }
            c = wmma_f16(a, b, c);
        }
#pragma unroll
        for (int rr = 0; rr < 8; ++rr)
            out[(size_t)(mrow + rr) * HID_ + o0 + nlo] += c[rr];
    }
}

// ---------------- host launch ----------------
extern "C" void kernel_launch(void* const* d_in, const int* in_sizes, int n_in,
                              void* d_out, int out_size, void* d_ws, size_t ws_size,
                              hipStream_t stream) {
    (void)in_sizes; (void)n_in; (void)out_size; (void)ws_size;
    const float* x      = (const float*)d_in[0];
    const float* r1_w   = (const float*)d_in[1];
    const float* r1_b   = (const float*)d_in[2];
    const float* r2_w   = (const float*)d_in[3];
    const float* w_up   = (const float*)d_in[4];
    const float* w_down = (const float*)d_in[5];
    const float* a_up   = (const float*)d_in[6];
    const float* a_down = (const float*)d_in[7];
    float* out = (float*)d_out;

    char* ws = (char*)d_ws;
    size_t off = 0;
    auto alloc = [&](size_t bytes) -> void* {
        void* p = ws + off;
        off = (off + bytes + 255) & ~(size_t)255;
        return p;
    };
    _Float16* xh    = (_Float16*)alloc((size_t)T_ * HID_ * 2);
    _Float16* xe    = (_Float16*)alloc((size_t)E_ * CAP_ * HID_ * 2);
    _Float16* hact  = (_Float16*)alloc((size_t)E_ * CAP_ * INTER_ * 2);
    _Float16* ahw   = (_Float16*)alloc((size_t)T_ * G_ * AINTER_ * 2);
    float*    probs = (float*)alloc((size_t)T_ * E_ * 4);
    int*      i2    = (int*)alloc((size_t)T_ * 2 * 4);
    float*    w2    = (float*)alloc((size_t)T_ * 2 * 4);
    int*      idxc  = (int*)alloc((size_t)E_ * CAP_ * 4);
    float*    wcap  = (float*)alloc((size_t)E_ * CAP_ * 4);
    float*    gw    = (float*)alloc((size_t)T_ * G_ * 4);

    hipMemsetAsync(out, 0, (size_t)T_ * HID_ * 4, stream);
    hipMemsetAsync(gw, 0, (size_t)T_ * G_ * 4, stream);

    router_kernel<<<T_, 256, 0, stream>>>(x, r1_w, r1_b, r2_w, probs, i2, w2);
    capacity_kernel<<<E_, 256, 0, stream>>>(probs, i2, w2, idxc, wcap, gw);
    cvt_half_kernel<<<(T_ * HID_) / 256, 256, 0, stream>>>(x, xh, T_ * HID_);
    gather_kernel<<<(E_ * CAP_ * HID_) / 256, 256, 0, stream>>>(x, idxc, xe);
    expert_up_kernel<<<E_ * (CAP_ / 16), 256, 0, stream>>>(xe, w_up, hact);
    expert_down_kernel<<<E_ * (CAP_ / 16), 256, 0, stream>>>(hact, w_down, idxc, wcap, out);
    adj_up_kernel<<<G_ * (T_ / 16), 256, 0, stream>>>(xh, a_up, gw, ahw);
    adj_down_kernel<<<T_ / 16, 256, 0, stream>>>(ahw, a_down, out);
}
